// TransformerModel_21680994910318
// MI455X (gfx1250) — matrix operand order repaired and checked by reading, not compiled, not run on hardware
//
#include <hip/hip_runtime.h>
#include <cstddef>
#include <cstdint>

// ---------------------------------------------------------------------------
// GPT-style transformer forward + CE loss for MI455X (gfx1250, wave32, WMMA).
// All matmuls run on v_wmma_f32_16x16x32_bf16 (bf16 in, f32 accumulate).
// GEMMs stage tiles through LDS; A-tile staging uses async global->LDS
// (gfx1250) when the builtin is available, plus global_prefetch for next tile.
// ---------------------------------------------------------------------------

#define DEV __device__ __forceinline__

typedef __attribute__((ext_vector_type(16))) __bf16 v16bf;
typedef __attribute__((ext_vector_type(8)))  __bf16 v8bf;
typedef __attribute__((ext_vector_type(8)))  float  v8f;
typedef __attribute__((ext_vector_type(4)))  int    v4i;

#if __has_builtin(__builtin_amdgcn_global_load_async_to_lds_b128)
#define HAVE_ASYNC_LDS 1
// async-to-LDS builtin takes (v4i AS1*, v4i AS3*, imm offset, imm cpol)
typedef __attribute__((address_space(1))) v4i gv4i;
typedef __attribute__((address_space(3))) v4i lv4i;
#else
#define HAVE_ASYNC_LDS 0
#endif

namespace cfg {
constexpr int B = 2, S = 2048, V = 32000, D = 1024, H = 16, L = 8;
constexpr int F = 4 * D;        // 4096
constexpr int DK = D / H;       // 64
constexpr int M = B * S;        // 4096 rows of the token stream
}

// ---- bf16 <-> f32 (bit-exact RNE, no reliance on __bf16 arithmetic) -------
DEV __bf16 f2bf(float f) {
  unsigned u = __float_as_uint(f);
  u += 0x7FFFu + ((u >> 16) & 1u);          // round-to-nearest-even
  unsigned short h = (unsigned short)(u >> 16);
  __bf16 r; __builtin_memcpy(&r, &h, 2); return r;
}
DEV float bf2f(__bf16 b) {
  unsigned short h; __builtin_memcpy(&h, &b, 2);
  return __uint_as_float(((unsigned)h) << 16);
}

// ---- WMMA wrapper ----------------------------------------------------------
DEV v8f wmma_bf16(v16bf a, v16bf b, v8f c) {
  // D = A(16x32) * B(32x16) + C, f32 accumulate
  return __builtin_amdgcn_wmma_f32_16x16x32_bf16(
      /*neg_a=*/false, a, /*neg_b=*/false, b,
      /*c_mod=*/(short)0, c, /*reuse_a=*/false, /*reuse_b=*/false);
}

// ---- Fragment loaders (contiguous k = 16*half + s, same order in A and B) ------------------
// A (16x32 bf16, row-major source, leading dim lda):
//   lane l: row = row0 + (l&15); half = l>>4
//   VGPR 0..7 : K = 16*half + {0..15}
DEV v16bf load_A_frag(const __bf16* A, int lda, int row0, int k0) {
  const int lane = threadIdx.x & 31;
  const int r    = row0 + (lane & 15);
  const int half = lane >> 4;
  const __bf16* p = A + (size_t)r * lda + k0 + 16 * half;
  v8bf lo = *(const v8bf*)(p);        // K = 16h .. 16h+7     (16B load)
  v8bf hi = *(const v8bf*)(p + 8);    // K = 16h+8 .. 16h+15  (16B load)
  return __builtin_shufflevector(lo, hi, 0,1,2,3,4,5,6,7,8,9,10,11,12,13,14,15);
}

// B (32x16 bf16): lane = N column; VGPR v holds K = 2v + 16*half (lo), +1 (hi)
// Generic row-major K x N source: column-strided scalar loads (used by loss).
DEV v16bf load_B_frag(const __bf16* Bm, int ldb, int k0, int n0) {
  const int lane = threadIdx.x & 31;
  const int n    = n0 + (lane & 15);
  const int half = lane >> 4;
  v16bf b;
  #pragma unroll
  for (int v = 0; v < 8; ++v) {
    int k = k0 + 2 * v + 16 * half;
    b[2*v]   = Bm[(size_t)k       * ldb + n];
    b[2*v+1] = Bm[(size_t)(k + 1) * ldb + n];
  }
  return b;
}

// B-frag from an N-major (transposed) tile: column n is contiguous in k.
DEV v16bf load_Bt_frag(const __bf16* Bt, int stride, int k0, int n0) {
  const int lane = threadIdx.x & 31;
  const int n    = n0 + (lane & 15);
  const int half = lane >> 4;
  const __bf16* p = Bt + (size_t)n * stride + k0 + 16 * half;
  v8bf lo = *(const v8bf*)(p);
  v8bf hi = *(const v8bf*)(p + 8);
  return __builtin_shufflevector(lo, hi, 0,1,2,3,4,5,6,7,8,9,10,11,12,13,14,15);
}

// B-frag for scores = Q * K^T : contract dim = dk, N = key position.
DEV v16bf load_KT_frag(const __bf16* Kh, int ldk, int kv0, int dk0) {
  const int lane = threadIdx.x & 31;
  int kv = kv0 + (lane & 15);
  if (kv > cfg::S - 1) kv = cfg::S - 1;       // clamp (masked to 0 by softmax)
  const int half = lane >> 4;
  const __bf16* p = Kh + (size_t)kv * ldk + dk0 + 16 * half;
  v8bf lo = *(const v8bf*)(p);
  v8bf hi = *(const v8bf*)(p + 8);
  return __builtin_shufflevector(lo, hi, 0,1,2,3,4,5,6,7,8,9,10,11,12,13,14,15);
}

// B-frag for O = P * V : contract dim = key position, N = dk.
DEV v16bf load_V_frag(const __bf16* Vh, int ldv, int kv0, int dk0) {
  const int lane = threadIdx.x & 31;
  const int n    = dk0 + (lane & 15);
  const int half = lane >> 4;
  v16bf b;
  #pragma unroll
  for (int v = 0; v < 8; ++v) {
    int k0i = kv0 + 2 * v + 16 * half, k1i = k0i + 1;
    if (k0i > cfg::S - 1) k0i = cfg::S - 1;
    if (k1i > cfg::S - 1) k1i = cfg::S - 1;
    b[2*v]   = Vh[(size_t)k0i * ldv + n];
    b[2*v+1] = Vh[(size_t)k1i * ldv + n];
  }
  return b;
}

// ---- in-16-lane-group butterfly reductions (wave32) ------------------------
DEV float rowmax16(float v) {
  #pragma unroll
  for (int m = 1; m < 16; m <<= 1) v = fmaxf(v, __shfl_xor(v, m, 32));
  return v;
}
DEV float rowsum16(float v) {
  #pragma unroll
  for (int m = 1; m < 16; m <<= 1) v += __shfl_xor(v, m, 32);
  return v;
}

DEV float gelu_exact(float x) {
  return 0.5f * x * (1.0f + erff(x * 0.70710678118654752440f));
}

// ===========================================================================
// Kernels
// ===========================================================================

__global__ void convert_kernel(const float* __restrict__ src,
                               __bf16* __restrict__ dst, size_t n) {
  size_t i = (size_t)blockIdx.x * blockDim.x + threadIdx.x;
  if (i < n) dst[i] = f2bf(src[i]);
}

__global__ void embed_kernel(const int* __restrict__ ids,
                             const float* __restrict__ tok,
                             const float* __restrict__ pos,
                             float* __restrict__ xf, __bf16* __restrict__ xb) {
  using namespace cfg;
  size_t idx = (size_t)blockIdx.x * blockDim.x + threadIdx.x;
  int m = (int)(idx / D), d = (int)(idx % D);
  int s = m % S;
  float v = tok[(size_t)ids[m] * D + d] + pos[(size_t)s * D + d];
  xf[idx] = v;
  xb[idx] = f2bf(v);
}

// ---------------------------------------------------------------------------
// C[M,N] = act(A[M,K] @ W[K,N] + bias).  ACT: 0 none, 1 exact GELU.
// Block 256 threads = 8 waves; block tile 128(M) x 64(N); K staged in LDS in
// steps of 64.  A tile staged with async global->LDS b128 when available;
// B tile staged transposed (N-major) so fragments are 16B contiguous reads.
// Per wave: 32x32 output (4 accumulators), 8 WMMA per K=64 stage.
// ---------------------------------------------------------------------------
template <int ACT>
__global__ __launch_bounds__(256)
void gemm_kernel(const __bf16* __restrict__ A, const __bf16* __restrict__ W,
                 const float* __restrict__ bias,
                 float* __restrict__ Cf, __bf16* __restrict__ Cb,
                 int Mi, int Ni, int Ki) {
  constexpr int BM = 128, BN = 64, BK = 64;
  constexpr int AST = BK + 8;   // padded LDS strides (still 16B aligned)
  constexpr int BST = BK + 8;
  __shared__ __align__(16) __bf16 As[BM * AST];   // ~18 KB
  __shared__ __align__(16) __bf16 Bt[BN * BST];   // ~9 KB, transposed (N-major)

  const int tid  = threadIdx.x;
  const int w    = tid >> 5;
  const int wm   = w & 3;            // 4 row groups of 32
  const int wn   = w >> 2;           // 2 col groups of 32
  const int rowb = blockIdx.y * BM;
  const int colb = blockIdx.x * BN;

  // staging decomposition
  const int rowA = tid >> 1;               // A row handled by this thread
  const int kcA  = (tid & 1) * 8;          // base k-chunk within row
  const int krB  = tid >> 3;               // first B row handled (of 2)
  const int ncB  = (tid & 7) * 8;          // 8-col chunk within B row

  v8f c00 = {}, c01 = {}, c10 = {}, c11 = {};

  for (int k0 = 0; k0 < Ki; k0 += BK) {
    // ---- stage A tile: 128 x 64 bf16 = 1024 x 16B chunks, 4 per thread ----
    #pragma unroll
    for (int i = 0; i < 2; ++i) {
      const int r   = rowA;                 // 0..127
      const int kc  = kcA + i * 16;         // {0,8} or {16,24}
      const int ko0 = kc;                   // chunk 1
      const int ko1 = kc + 32;              // chunk 2
      const __bf16* g0 = A + (size_t)(rowb + r) * Ki + k0 + ko0;
      const __bf16* g1 = A + (size_t)(rowb + r) * Ki + k0 + ko1;
      __bf16* l0 = &As[r * AST + ko0];
      __bf16* l1 = &As[r * AST + ko1];
#if HAVE_ASYNC_LDS
      __builtin_amdgcn_global_load_async_to_lds_b128((gv4i*)g0, (lv4i*)l0, 0, 0);
      __builtin_amdgcn_global_load_async_to_lds_b128((gv4i*)g1, (lv4i*)l1, 0, 0);
#else
      *(v8bf*)l0 = *(const v8bf*)g0;
      *(v8bf*)l1 = *(const v8bf*)g1;
#endif
    }
    // ---- stage B tile transposed: 64 x 64, 2 row-chunks per thread --------
    #pragma unroll
    for (int i = 0; i < 2; ++i) {
      const int kr = krB + i * 32;          // 0..63
      v8bf bv = *(const v8bf*)(W + (size_t)(k0 + kr) * Ni + colb + ncB);
      #pragma unroll
      for (int j = 0; j < 8; ++j) Bt[(ncB + j) * BST + kr] = bv[j];
    }
    // prefetch next K tile into cache while we compute this one
    if (k0 + BK < Ki) {
      __builtin_prefetch(A + (size_t)(rowb + rowA) * Ki + k0 + BK + kcA, 0, 0);
      __builtin_prefetch(W + (size_t)(k0 + BK + krB) * Ni + colb + ncB, 0, 0);
    }
#if HAVE_ASYNC_LDS
#if __has_builtin(__builtin_amdgcn_s_wait_asynccnt)
    __builtin_amdgcn_s_wait_asynccnt(0);
#else
    asm volatile("s_wait_asynccnt 0x0" ::: "memory");
#endif
#endif
    __syncthreads();

    // ---- compute: 2 K-substeps x 4 WMMA -----------------------------------
    #pragma unroll
    for (int ks = 0; ks < BK; ks += 32) {
      v16bf a0 = load_A_frag(As, AST, wm * 32,      ks);
      v16bf a1 = load_A_frag(As, AST, wm * 32 + 16, ks);
      v16bf b0 = load_Bt_frag(Bt, BST, ks, wn * 32);
      v16bf b1 = load_Bt_frag(Bt, BST, ks, wn * 32 + 16);
      c00 = wmma_bf16(a0, b0, c00);
      c01 = wmma_bf16(a0, b1, c01);
      c10 = wmma_bf16(a1, b0, c10);
      c11 = wmma_bf16(a1, b1, c11);
    }
    __syncthreads();
  }

  // ---- epilogue: bias (+GELU), store f32 and/or bf16 ----------------------
  const int lane = tid & 31;
  const int cidx = lane & 15, half = lane >> 4;
  #pragma unroll
  for (int ti = 0; ti < 2; ++ti) {
    #pragma unroll
    for (int tj = 0; tj < 2; ++tj) {
      v8f acc = (ti == 0) ? (tj == 0 ? c00 : c01) : (tj == 0 ? c10 : c11);
      const int n = colb + wn * 32 + tj * 16 + cidx;
      const float bia = bias ? bias[n] : 0.f;
      #pragma unroll
      for (int r = 0; r < 8; ++r) {
        const int m = rowb + wm * 32 + ti * 16 + r + 8 * half;
        float v = acc[r] + bia;
        if (ACT == 1) v = gelu_exact(v);
        if (Cf) Cf[(size_t)m * Ni + n] = v;
        if (Cb) Cb[(size_t)m * Ni + n] = f2bf(v);
      }
    }
  }
}

// Flash attention, causal.  One wave per (b, h, 16-row q tile).
__global__ __launch_bounds__(32)
void attn_kernel(const __bf16* __restrict__ Qb, const __bf16* __restrict__ Kb,
                 const __bf16* __restrict__ Vb, __bf16* __restrict__ Ob) {
  using namespace cfg;
  const int qt = blockIdx.x, h = blockIdx.y, b = blockIdx.z;
  const int row0 = qt * 16;
  const size_t base = (size_t)b * S * D + (size_t)h * DK;
  const __bf16* Qh = Qb + base;
  const __bf16* Kh = Kb + base;
  const __bf16* Vh = Vb + base;

  const int lane = threadIdx.x & 31;
  const int cidx = lane & 15, half = lane >> 4;

  v16bf qa0 = load_A_frag(Qh, D, row0, 0);    // dk 0..31
  v16bf qa1 = load_A_frag(Qh, D, row0, 32);   // dk 32..63

  v8f o0 = {}, o1 = {}, o2 = {}, o3 = {};
  float mrow[8], lrow[8];
  #pragma unroll
  for (int r = 0; r < 8; ++r) { mrow[r] = -1e30f; lrow[r] = 0.f; }

  __shared__ __align__(16) __bf16 Pl[16 * 32];  // P tile re-layout C-frag->A-frag
  const float scale = 0.125f;                   // 1/sqrt(64)

  for (int kb = 0; kb <= row0; kb += 32) {      // causal: only kv <= row0+15
    v8f s0 = {}, s1 = {};
    {
      v16bf k00 = load_KT_frag(Kh, D, kb, 0);
      v16bf k01 = load_KT_frag(Kh, D, kb, 32);
      s0 = wmma_bf16(qa0, k00, s0);
      s0 = wmma_bf16(qa1, k01, s0);
      v16bf k10 = load_KT_frag(Kh, D, kb + 16, 0);
      v16bf k11 = load_KT_frag(Kh, D, kb + 16, 32);
      s1 = wmma_bf16(qa0, k10, s1);
      s1 = wmma_bf16(qa1, k11, s1);
    }
    const int kv0 = kb + cidx, kv1 = kb + 16 + cidx;
    #pragma unroll
    for (int r = 0; r < 8; ++r) {
      int m = row0 + r + 8 * half;
      float a0 = s0[r] * scale;
      float a1 = s1[r] * scale;
      if (kv0 > m) a0 = -1e9f;                  // causal mask (matches ref)
      if (kv1 > m) a1 = -1e9f;
      float tm = rowmax16(fmaxf(a0, a1));
      float nm = fmaxf(mrow[r], tm);
      float alpha = __expf(mrow[r] - nm);
      mrow[r] = nm;
      float p0 = __expf(a0 - nm);
      float p1 = __expf(a1 - nm);
      lrow[r] = lrow[r] * alpha + rowsum16(p0 + p1);
      o0[r] *= alpha; o1[r] *= alpha; o2[r] *= alpha; o3[r] *= alpha;
      Pl[(r + 8 * half) * 32 + cidx]      = f2bf(p0);
      Pl[(r + 8 * half) * 32 + cidx + 16] = f2bf(p1);
    }
    __syncthreads();
    v16bf pa  = load_A_frag(Pl, 32, 0, 0);      // P as A-frag (16x32 bf16)
    v16bf vb0 = load_V_frag(Vh, D, kb, 0);
    v16bf vb1 = load_V_frag(Vh, D, kb, 16);
    v16bf vb2 = load_V_frag(Vh, D, kb, 32);
    v16bf vb3 = load_V_frag(Vh, D, kb, 48);
    o0 = wmma_bf16(pa, vb0, o0);
    o1 = wmma_bf16(pa, vb1, o1);
    o2 = wmma_bf16(pa, vb2, o2);
    o3 = wmma_bf16(pa, vb3, o3);
    __syncthreads();
  }
  #pragma unroll
  for (int j = 0; j < 4; ++j) {
    v8f o = (j == 0) ? o0 : (j == 1) ? o1 : (j == 2) ? o2 : o3;
    #pragma unroll
    for (int r = 0; r < 8; ++r) {
      int m  = row0 + r + 8 * half;
      int dk = 16 * j + cidx;
      Ob[(size_t)(b * S + m) * D + h * DK + dk] = f2bf(o[r] / lrow[r]);
    }
  }
}

// Residual add + LayerNorm; writes f32 stream + bf16 copy.  One block per row.
__global__ __launch_bounds__(256)
void ln_kernel(float* __restrict__ x, const float* __restrict__ delta,
               const float* __restrict__ sc, const float* __restrict__ bi,
               __bf16* __restrict__ xb) {
  using namespace cfg;
  __shared__ float red[256];
  const int row = blockIdx.x, tid = threadIdx.x;
  float v[4];
  float s = 0.f;
  #pragma unroll
  for (int i = 0; i < 4; ++i) {
    int idx = i * 256 + tid;
    float y = x[(size_t)row * D + idx];
    if (delta) y += delta[(size_t)row * D + idx];
    v[i] = y; s += y;
  }
  red[tid] = s; __syncthreads();
  for (int off = 128; off > 0; off >>= 1) {
    if (tid < off) red[tid] += red[tid + off];
    __syncthreads();
  }
  float mu = red[0] * (1.f / D);
  __syncthreads();
  s = 0.f;
  #pragma unroll
  for (int i = 0; i < 4; ++i) { float d = v[i] - mu; s += d * d; }
  red[tid] = s; __syncthreads();
  for (int off = 128; off > 0; off >>= 1) {
    if (tid < off) red[tid] += red[tid + off];
    __syncthreads();
  }
  float rstd = rsqrtf(red[0] * (1.f / D) + 1e-5f);
  #pragma unroll
  for (int i = 0; i < 4; ++i) {
    int idx = i * 256 + tid;
    float o = (v[i] - mu) * rstd * sc[idx] + bi[idx];
    x[(size_t)row * D + idx]  = o;
    xb[(size_t)row * D + idx] = f2bf(o);
  }
}

// Fused LM-head + streaming log-softmax CE.  One block per 16-token tile.
// 8 waves sweep the 2000 vocab tiles (16 cols each) with online max/sumexp.
__global__ __launch_bounds__(256)
void lm_loss_kernel(const __bf16* __restrict__ xb, const __bf16* __restrict__ Wlm,
                    const int* __restrict__ labels, float* __restrict__ nll) {
  using namespace cfg;
  __shared__ __align__(16) __bf16 Xs[16 * D];   // 32 KB staged activations
  __shared__ float RM[8][16], RS[8][16], RL[8][16];
  const int m0 = blockIdx.x * 16;
  {
    const v8bf* src = (const v8bf*)(xb + (size_t)m0 * D);
    v8bf* dst = (v8bf*)Xs;
    for (int i = threadIdx.x; i < 16 * D / 8; i += 256) dst[i] = src[i];
  }
  __syncthreads();
  const int w    = threadIdx.x >> 5;
  const int lane = threadIdx.x & 31;
  const int cidx = lane & 15, half = lane >> 4;

  float mx[8], sm[8], lv[8]; int labr[8];
  #pragma unroll
  for (int r = 0; r < 8; ++r) {
    mx[r] = -3.0e38f; sm[r] = 0.f; lv[r] = -3.0e38f;
    int gm = m0 + r + 8 * half;
    labr[r] = ((gm % S) == S - 1) ? -1 : labels[gm + 1];  // shifted labels
  }
  for (int t = w; t < V / 16; t += 8) {
    const int n0 = t * 16;
    v8f acc = {};
    for (int k = 0; k < D; k += 32) {
      v16bf a = load_A_frag(Xs, D, 0, k);
      v16bf b = load_B_frag(Wlm, V, k, n0);
      acc = wmma_bf16(a, b, acc);
    }
    const int n = n0 + cidx;
    #pragma unroll
    for (int r = 0; r < 8; ++r) {
      float val = acc[r];
      if (n == labr[r]) lv[r] = val;
      if (val > mx[r]) { sm[r] = sm[r] * __expf(mx[r] - val) + 1.f; mx[r] = val; }
      else             { sm[r] += __expf(val - mx[r]); }
    }
  }
  // reduce within 16-lane group (butterfly), combining (max, sumexp)
  #pragma unroll
  for (int r = 0; r < 8; ++r) {
    #pragma unroll
    for (int msk = 1; msk < 16; msk <<= 1) {
      float om = __shfl_xor(mx[r], msk, 32);
      float os = __shfl_xor(sm[r], msk, 32);
      float nm = fmaxf(mx[r], om);
      sm[r] = sm[r] * __expf(mx[r] - nm) + os * __expf(om - nm);
      mx[r] = nm;
      lv[r] = fmaxf(lv[r], __shfl_xor(lv[r], msk, 32));
    }
  }
  if (cidx == 0) {
    #pragma unroll
    for (int r = 0; r < 8; ++r) {
      int row = r + 8 * half;
      RM[w][row] = mx[r]; RS[w][row] = sm[r]; RL[w][row] = lv[r];
    }
  }
  __syncthreads();
  if (threadIdx.x < 16) {
    int row = threadIdx.x;
    float m = RM[0][row], s2 = RS[0][row], l2 = RL[0][row];
    #pragma unroll
    for (int ww = 1; ww < 8; ++ww) {
      float wm = RM[ww][row], wsv = RS[ww][row];
      float nm = fmaxf(m, wm);
      s2 = s2 * __expf(m - nm) + wsv * __expf(wm - nm);
      m = nm;
      l2 = fmaxf(l2, RL[ww][row]);
    }
    int gm = m0 + row;
    nll[gm] = ((gm % S) == S - 1) ? 0.f : (m + logf(s2) - l2);
  }
}

__global__ void reduce_kernel(const float* __restrict__ nll, float* __restrict__ out,
                              int n, float scale) {
  __shared__ float sd[256];
  float s = 0.f;
  for (int i = threadIdx.x; i < n; i += 256) s += nll[i];
  sd[threadIdx.x] = s; __syncthreads();
  for (int off = 128; off > 0; off >>= 1) {
    if (threadIdx.x < off) sd[threadIdx.x] += sd[threadIdx.x + off];
    __syncthreads();
  }
  if (threadIdx.x == 0) out[0] = sd[0] * scale;
}

// ===========================================================================
// Host launcher
// ===========================================================================
extern "C" void kernel_launch(void* const* d_in, const int* in_sizes, int n_in,
                              void* d_out, int out_size, void* d_ws, size_t ws_size,
                              hipStream_t stream) {
  using namespace cfg;
  (void)in_sizes; (void)n_in; (void)out_size; (void)ws_size;

  const int*   ids    = (const int*)d_in[0];
  const int*   labels = (const int*)d_in[1];
  const float* tok    = (const float*)d_in[2];
  const float* pos    = (const float*)d_in[3];
  const float* wq  = (const float*)d_in[4];  const float* bq = (const float*)d_in[5];
  const float* wk  = (const float*)d_in[6];  const float* bk = (const float*)d_in[7];
  const float* wv  = (const float*)d_in[8];  const float* bv = (const float*)d_in[9];
  const float* wo  = (const float*)d_in[10]; const float* bo = (const float*)d_in[11];
  const float* w1  = (const float*)d_in[12]; const float* b1 = (const float*)d_in[13];
  const float* w2  = (const float*)d_in[14]; const float* b2 = (const float*)d_in[15];
  const float* ln1s = (const float*)d_in[16]; const float* ln1b = (const float*)d_in[17];
  const float* ln2s = (const float*)d_in[18]; const float* ln2b = (const float*)d_in[19];
  const float* lnfs = (const float*)d_in[20]; const float* lnfb = (const float*)d_in[21];
  const float* lmw  = (const float*)d_in[22];

  // ---- workspace bump allocator (~370 MB) ----
  char* wsp = (char*)d_ws;
  auto alloc = [&](size_t bytes) -> void* {
    void* p = (void*)wsp; wsp += (bytes + 255) & ~(size_t)255; return p;
  };
  float*  xf   = (float*) alloc((size_t)M * D * 4);
  __bf16* xb   = (__bf16*)alloc((size_t)M * D * 2);
  __bf16* qb   = (__bf16*)alloc((size_t)M * D * 2);
  __bf16* kbuf = (__bf16*)alloc((size_t)M * D * 2);
  __bf16* vbuf = (__bf16*)alloc((size_t)M * D * 2);
  __bf16* aob  = (__bf16*)alloc((size_t)M * D * 2);
  float*  tmpf = (float*) alloc((size_t)M * D * 4);
  __bf16* hb   = (__bf16*)alloc((size_t)M * F * 2);
  __bf16* wqb  = (__bf16*)alloc((size_t)L * D * D * 2);
  __bf16* wkb  = (__bf16*)alloc((size_t)L * D * D * 2);
  __bf16* wvb  = (__bf16*)alloc((size_t)L * D * D * 2);
  __bf16* wob  = (__bf16*)alloc((size_t)L * D * D * 2);
  __bf16* w1b  = (__bf16*)alloc((size_t)L * D * F * 2);
  __bf16* w2b  = (__bf16*)alloc((size_t)L * F * D * 2);
  __bf16* lmb  = (__bf16*)alloc((size_t)D * V * 2);
  float*  nll  = (float*) alloc((size_t)M * 4);

  auto cvt = [&](const float* src, __bf16* dst, size_t n) {
    convert_kernel<<<dim3((unsigned)((n + 255) / 256)), 256, 0, stream>>>(src, dst, n);
  };
  cvt(wq, wqb, (size_t)L * D * D);
  cvt(wk, wkb, (size_t)L * D * D);
  cvt(wv, wvb, (size_t)L * D * D);
  cvt(wo, wob, (size_t)L * D * D);
  cvt(w1, w1b, (size_t)L * D * F);
  cvt(w2, w2b, (size_t)L * F * D);
  cvt(lmw, lmb, (size_t)D * V);

  embed_kernel<<<dim3((M * D) / 256), 256, 0, stream>>>(ids, tok, pos, xf, xb);

  const dim3 gDD(D / 64, M / 128);   // N=1024 GEMMs, 128-row block tiles
  const dim3 gDF(F / 64, M / 128);   // N=4096 GEMM (fc1)
  for (int l = 0; l < L; ++l) {
    const size_t oDD = (size_t)l * D * D;
    const size_t oDF = (size_t)l * D * F;
    gemm_kernel<0><<<gDD, 256, 0, stream>>>(xb, wqb + oDD, bq + (size_t)l * D, nullptr, qb,   M, D, D);
    gemm_kernel<0><<<gDD, 256, 0, stream>>>(xb, wkb + oDD, bk + (size_t)l * D, nullptr, kbuf, M, D, D);
    gemm_kernel<0><<<gDD, 256, 0, stream>>>(xb, wvb + oDD, bv + (size_t)l * D, nullptr, vbuf, M, D, D);
    attn_kernel<<<dim3(S / 16, H, B), 32, 0, stream>>>(qb, kbuf, vbuf, aob);
    gemm_kernel<0><<<gDD, 256, 0, stream>>>(aob, wob + oDD, bo + (size_t)l * D, tmpf, nullptr, M, D, D);
    ln_kernel<<<M, 256, 0, stream>>>(xf, tmpf, ln1s + (size_t)l * D, ln1b + (size_t)l * D, xb);
    gemm_kernel<1><<<gDF, 256, 0, stream>>>(xb, w1b + oDF, b1 + (size_t)l * F, nullptr, hb, M, F, D);
    gemm_kernel<0><<<gDD, 256, 0, stream>>>(hb, w2b + oDF, b2 + (size_t)l * D, tmpf, nullptr, M, D, F);
    ln_kernel<<<M, 256, 0, stream>>>(xf, tmpf, ln2s + (size_t)l * D, ln2b + (size_t)l * D, xb);
  }
  ln_kernel<<<M, 256, 0, stream>>>(xf, nullptr, lnfs, lnfb, xb);
  lm_loss_kernel<<<M / 16, 256, 0, stream>>>(xb, lmb, labels, nll);
  reduce_kernel<<<1, 256, 0, stream>>>(nll, (float*)d_out, M, 1.0f / (float)(B * (S - 1)));
}
